// QuantLinearGPTQ_5033701671375
// MI455X (gfx1250) — compile-verified, matrix operand order checked by reading, not run on hardware
//
#include <hip/hip_runtime.h>
#include <hip/hip_bf16.h>

typedef __attribute__((ext_vector_type(16))) _Float16 v16h;
typedef __attribute__((ext_vector_type(8)))  _Float16 v8h;
typedef __attribute__((ext_vector_type(4)))  _Float16 v4h;
typedef __attribute__((ext_vector_type(8)))  float    v8f;
typedef __attribute__((ext_vector_type(4)))  float    v4f;
typedef __attribute__((ext_vector_type(4)))  unsigned int u32x4;
typedef __attribute__((ext_vector_type(8)))  int      i32x8;
typedef __attribute__((ext_vector_type(4)))  int      i32x4;

#define M_DIM 128
#define K_DIM 4096
#define N_DIM 12288
#define QZ_COLS (N_DIM / 8)
#define BK 64                  // K-chunk staged in LDS
#define LDS_STRIDE 72          // halfs per row (144 B; TDM pads 64->72)
#define BUF_HALFS (M_DIM * LDS_STRIDE)

// ---------------------------------------------------------------------------
// Pre-pass: x (f32) -> xh (f16), once.  1 MB written; lets the main kernel
// DMA f16 rows straight into LDS with the TDM (no conversion en route).
// ---------------------------------------------------------------------------
__global__ __launch_bounds__(256)
void cvt_x_to_f16(const float* __restrict__ x, _Float16* __restrict__ xh)
{
    const int i = (blockIdx.x * 256 + threadIdx.x) * 8;
    const v4f* s = (const v4f*)(x + i);
    const v4h a = __builtin_convertvector(s[0], v4h);
    const v4h b = __builtin_convertvector(s[1], v4h);
    *(v8h*)(xh + i) = __builtin_shufflevector(a, b, 0,1,2,3,4,5,6,7);
}

// ---------------------------------------------------------------------------
// TDM: DMA a 64x128 f16 tile (rows of 64 halfs, stride 4096) into LDS,
// padding +4 DWORDs every 32 DWORDs so LDS rows land at stride 144 B
// (bank-conflict-free for the 16-lane ds_load_b128 A-fragment pattern).
// D# packing per CDNA5 ISA §8.3-8.5.  clang-23 6-arg builtin form.
// ---------------------------------------------------------------------------
__device__ __forceinline__
void tdm_load_chunk(const _Float16* gsrc, unsigned lds_byte_addr)
{
    const unsigned long long ga = (unsigned long long)(uintptr_t)gsrc;
    u32x4 g0;
    g0.x = 1u;                                    // count=1, user mode
    g0.y = lds_byte_addr;                         // lds_addr
    g0.z = (unsigned)ga;                          // global_addr[31:0]
    g0.w = (unsigned)((ga >> 32) & 0x1FFFFFFu)    // global_addr[56:32]
         | (2u << 30);                            // type=2 ("image")
    i32x8 g1;
    g1[0] = (1 << 16)      // data_size = 2 bytes
          | (1 << 20)      // pad_enable
          | (4 << 22)      // pad_interval: every 32 DWORDs (128 B)
          | (3 << 25);     // pad_amount: 4 DWORDs (16 B)
    g1[1] = (int)(4096u << 16);   // tensor_dim0 = 4096 (low16 in [31:16])
    g1[2] = (int)(128u  << 16);   // tensor_dim0 hi=0 | tensor_dim1 = 128
    g1[3] = (int)(64u   << 16);   // tensor_dim1 hi=0 | tile_dim0 = 64
    g1[4] = 128;                  // tile_dim1 = 128, tile_dim2 = 0
    g1[5] = 4096;                 // tensor_dim0_stride = 4096 (elements)
    g1[6] = 0;
    g1[7] = 0;
    const i32x4 z4 = (i32x4)0;    // groups 2/3 unused (2-D tensor)
    const i32x8 z8 = (i32x8)0;
    __builtin_amdgcn_tensor_load_to_lds(g0, g1, z4, z4, z8, 0);
}

// ---------------------------------------------------------------------------
// Main GEMM: per block 128(M) x 128(N); 8 waves, one N=16 strip each,
// 8 M-tiles per wave; K in 64-chunks double-buffered in LDS via TDM.
// ---------------------------------------------------------------------------
__global__ __launch_bounds__(256)
void gptq_gemm_wmma_tdm(const _Float16* __restrict__ xh,
                        const int*   __restrict__ qweight,
                        const int*   __restrict__ qzeros,
                        const float* __restrict__ scales,
                        const float* __restrict__ bias,
                        float* __restrict__ out)
{
    __shared__ _Float16 lds_a[2 * BUF_HALFS];     // 2 x 18432 B

    const int tid  = threadIdx.x;
    const int wave = tid >> 5;
    const int lane = tid & 31;
    const int col  = lane & 15;
    const int hi   = lane >> 4;

    const int n = blockIdx.x * 128 + wave * 16 + col;

    v8f acc[8];
    #pragma unroll
    for (int t = 0; t < 8; ++t) acc[t] = (v8f)0.0f;

    const float bias_n = bias[n];
    const unsigned lds_base = (unsigned)(uintptr_t)(&lds_a[0]);

    // prologue: DMA chunk 0 into buffer 0
    if (wave == 0) {
        tdm_load_chunk(xh, lds_base);
        __builtin_amdgcn_s_wait_tensorcnt(0);
    }
    __syncthreads();

    for (int ko = 0; ko < K_DIM; ko += BK) {
        const int buf = (ko / BK) & 1;

        // kick off next chunk's DMA into the other buffer (overlaps compute)
        if (ko + BK < K_DIM && wave == 0)
            tdm_load_chunk(xh + (ko + BK),
                           lds_base + (buf ^ 1) * (BUF_HALFS * 2));

        // ---- hoist the 4 packed-weight dwords for this 64-K chunk (NT) ----
        const int rbase = (ko >> 3) + hi;
        unsigned q[4];
        q[0] = (unsigned)__builtin_nontemporal_load(&qweight[(rbase    ) * N_DIM + n]);
        q[1] = (unsigned)__builtin_nontemporal_load(&qweight[(rbase + 2) * N_DIM + n]);
        q[2] = (unsigned)__builtin_nontemporal_load(&qweight[(rbase + 4) * N_DIM + n]);
        q[3] = (unsigned)__builtin_nontemporal_load(&qweight[(rbase + 6) * N_DIM + n]);
        if (ko + BK < K_DIM)
            __builtin_prefetch(&qweight[(rbase + 8) * N_DIM + n], 0, 0);

        // ---- per-group dequant constants (group = 128 K) ----
        const int g = ko >> 7;
        const float s = scales[g * N_DIM + n];
        const unsigned zq = (unsigned)qzeros[g * QZ_COLS + (n >> 3)];
        const float nsz = -s * (float)((zq >> ((n & 7) * 4)) & 0xF);

        const _Float16* abuf = lds_a + buf * BUF_HALFS;

        #pragma unroll
        for (int step = 0; step < 2; ++step) {
            const int kk = step * 32;
            const unsigned q0 = q[step * 2];
            const unsigned q1 = q[step * 2 + 1];

            // dequantize 16 nibbles -> f16 B fragment
            v16h bfrag;
            #pragma unroll
            for (int i = 0; i < 8; ++i) {
                const float w0 = (float)((q0 >> (4 * i)) & 0xF);
                const float w1 = (float)((q1 >> (4 * i)) & 0xF);
                bfrag[i]     = (_Float16)__builtin_fmaf(w0, s, nsz);
                bfrag[i + 8] = (_Float16)__builtin_fmaf(w1, s, nsz);
            }

            // preload all 8 A-fragments, then 8 back-to-back WMMAs
            const int a_half_off = kk + hi * 8;
            v16h afrag[8];
            #pragma unroll
            for (int t = 0; t < 8; ++t) {
                const int m = t * 16 + col;
                const v8h* ap = (const v8h*)&abuf[m * LDS_STRIDE + a_half_off];
                const v8h alo = ap[0];
                const v8h ahi = ap[2];
                afrag[t] = __builtin_shufflevector(
                    alo, ahi, 0,1,2,3,4,5,6,7,8,9,10,11,12,13,14,15);
            }
            #pragma unroll
            for (int t = 0; t < 8; ++t) {
                acc[t] = __builtin_amdgcn_wmma_f32_16x16x32_f16(
                    false, afrag[t], false, bfrag,
                    (short)0, acc[t], false, false);
            }
        }

        // handoff: DMA done + everyone finished reading current buffer
        if (ko + BK < K_DIM) {
            if (wave == 0) __builtin_amdgcn_s_wait_tensorcnt(0);
            __syncthreads();
        }
    }

    // ---- store C + bias (write-once stream: non-temporal) ----
    #pragma unroll
    for (int t = 0; t < 8; ++t) {
        #pragma unroll
        for (int j = 0; j < 8; ++j) {
            const int m = t * 16 + hi * 8 + j;
            __builtin_nontemporal_store(acc[t][j] + bias_n, &out[m * N_DIM + n]);
        }
    }
}

// ---------------------------------------------------------------------------
// Fallback (no workspace): round-2 kernel, f32 x staged through VGPRs.
// ---------------------------------------------------------------------------
__global__ __launch_bounds__(256)
void gptq_gemm_wmma_fb(const float* __restrict__ x,
                       const int*   __restrict__ qweight,
                       const int*   __restrict__ qzeros,
                       const float* __restrict__ scales,
                       const float* __restrict__ bias,
                       float* __restrict__ out)
{
    __shared__ _Float16 lds_a[BUF_HALFS];

    const int tid  = threadIdx.x;
    const int wave = tid >> 5;
    const int lane = tid & 31;
    const int col  = lane & 15;
    const int hi   = lane >> 4;
    const int n = blockIdx.x * 128 + wave * 16 + col;

    v8f acc[8];
    #pragma unroll
    for (int t = 0; t < 8; ++t) acc[t] = (v8f)0.0f;
    const float bias_n = bias[n];

    const int row_ld  = tid >> 1;
    const int koff_ld = (tid & 1) * 32;

    for (int ko = 0; ko < K_DIM; ko += BK) {
        {
            const v4f* src = (const v4f*)(x + row_ld * K_DIM + ko + koff_ld);
            v4h* dst = (v4h*)&lds_a[row_ld * LDS_STRIDE + koff_ld];
            #pragma unroll
            for (int i = 0; i < 8; ++i)
                dst[i] = __builtin_convertvector(src[i], v4h);
        }
        __syncthreads();

        const int rbase = (ko >> 3) + hi;
        unsigned q[4];
        q[0] = (unsigned)qweight[(rbase    ) * N_DIM + n];
        q[1] = (unsigned)qweight[(rbase + 2) * N_DIM + n];
        q[2] = (unsigned)qweight[(rbase + 4) * N_DIM + n];
        q[3] = (unsigned)qweight[(rbase + 6) * N_DIM + n];

        const int g = ko >> 7;
        const float s = scales[g * N_DIM + n];
        const unsigned zq = (unsigned)qzeros[g * QZ_COLS + (n >> 3)];
        const float nsz = -s * (float)((zq >> ((n & 7) * 4)) & 0xF);

        #pragma unroll
        for (int step = 0; step < 2; ++step) {
            const int kk = step * 32;
            const unsigned q0 = q[step * 2], q1 = q[step * 2 + 1];
            v16h bfrag;
            #pragma unroll
            for (int i = 0; i < 8; ++i) {
                bfrag[i]     = (_Float16)__builtin_fmaf((float)((q0 >> (4*i)) & 0xF), s, nsz);
                bfrag[i + 8] = (_Float16)__builtin_fmaf((float)((q1 >> (4*i)) & 0xF), s, nsz);
            }
            const int a_half_off = kk + hi * 8;
            v16h afrag[8];
            #pragma unroll
            for (int t = 0; t < 8; ++t) {
                const v8h* ap = (const v8h*)&lds_a[(t*16 + col) * LDS_STRIDE + a_half_off];
                afrag[t] = __builtin_shufflevector(
                    ap[0], ap[2], 0,1,2,3,4,5,6,7,8,9,10,11,12,13,14,15);
            }
            #pragma unroll
            for (int t = 0; t < 8; ++t)
                acc[t] = __builtin_amdgcn_wmma_f32_16x16x32_f16(
                    false, afrag[t], false, bfrag, (short)0, acc[t], false, false);
        }
        __syncthreads();
    }

    #pragma unroll
    for (int t = 0; t < 8; ++t)
        #pragma unroll
        for (int j = 0; j < 8; ++j)
            out[(t*16 + hi*8 + j) * N_DIM + n] = acc[t][j] + bias_n;
}

extern "C" void kernel_launch(void* const* d_in, const int* in_sizes, int n_in,
                              void* d_out, int out_size, void* d_ws, size_t ws_size,
                              hipStream_t stream) {
    const float* x       = (const float*)d_in[0];
    const int*   qweight = (const int*)  d_in[1];
    const int*   qzeros  = (const int*)  d_in[2];
    const float* scales  = (const float*)d_in[3];
    const float* bias    = (const float*)d_in[4];
    float* out = (float*)d_out;

    const size_t need = (size_t)M_DIM * K_DIM * sizeof(_Float16);   // 1 MB
    if (ws_size >= need) {
        _Float16* xh = (_Float16*)d_ws;
        cvt_x_to_f16<<<dim3((M_DIM * K_DIM) / (256 * 8)), dim3(256), 0, stream>>>(x, xh);
        gptq_gemm_wmma_tdm<<<dim3(N_DIM / 128), dim3(256), 0, stream>>>(
            xh, qweight, qzeros, scales, bias, out);
    } else {
        gptq_gemm_wmma_fb<<<dim3(N_DIM / 128), dim3(256), 0, stream>>>(
            x, qweight, qzeros, scales, bias, out);
    }
}
